// GroupedMLP_64750926954712
// MI455X (gfx1250) — matrix operand order repaired and checked by reading, not compiled, not run on hardware
//
#include <hip/hip_runtime.h>
#include <hip/hip_bf16.h>

// GroupedMLP for MI455X (gfx1250, wave32, WMMA).
// E=8 experts, H=2048, F=4096, T=16384 (2048 tokens/expert, uniform split —
// tokens_per_expert input is the constant T/E).
// f32 inputs converted to bf16 during LDS staging; f32 accumulation via
// v_wmma_f32_16x16x32_bf16; hidden intermediate kept bf16 in d_ws; kernel 2
// stages the bf16 hidden tile via global_load_async_to_lds_b128 (ASYNCcnt).

typedef __attribute__((ext_vector_type(16))) __bf16 v16bf;
typedef __attribute__((ext_vector_type(8)))  __bf16 v8bf;
typedef __attribute__((ext_vector_type(4)))  __bf16 v4bf;
typedef __attribute__((ext_vector_type(2)))  __bf16 v2bf;
typedef __attribute__((ext_vector_type(8)))  float  v8f;
typedef __attribute__((ext_vector_type(4)))  float  v4f;

namespace {
constexpr int kE   = 8;
constexpr int kH   = 2048;
constexpr int kF   = 4096;
constexpr int kTPE = 2048;   // tokens per expert

constexpr int BM = 256;      // block tile rows (tokens): 4 waves x 64
constexpr int BN = 64;       // block tile cols:          2 waves x 32
constexpr int BK = 32;       // K step (matches WMMA K for bf16)
constexpr int WM = 64;       // per-wave M (4 x 16 subtiles)
}

union AFrag { v16bf v; v8bf h[2]; };

// A-fragment (16-bit A 16x32, ISA 7.12.2): row m = lane&15;
// lanes<16 hold K 0..7 (VGPR0-3) and 16..23 (VGPR4-7); lanes>=16 hold 8..15 / 24..31.
__device__ inline v16bf load_a_frag(const __bf16* lds /*[..][BK] row-major*/,
                                    int mrow, int lane) {
  AFrag a;
  const int k0 = (lane < 16) ? 0 : 8;
  const __bf16* p = lds + mrow * BK;
  a.h[0] = *(const v8bf*)(p + k0);
  a.h[1] = *(const v8bf*)(p + k0 + 16);
  return a.v;
}

__device__ inline float silu_fast(float g) {
  // silu(g) = g * sigmoid(g); v_rcp_f32 precision is ample for bf16 output.
  return g * __builtin_amdgcn_rcpf(1.0f + __expf(-g));
}

// ---------------------------------------------------------------------------
// Kernel 1: hidden[e,t,f] = silu(x@Wg^T) * (x@Wu^T), output bf16.
// ---------------------------------------------------------------------------
__global__ __launch_bounds__(256) void gmlp_gateup_silu(
    const float* __restrict__ x,          // [E*TPE, H]
    const float* __restrict__ wgu,        // [E, 2F, H]
    __hip_bfloat16* __restrict__ hidden)  // [E, TPE, F]
{
  const int e  = blockIdx.z;
  const int mb = blockIdx.y * BM;   // token row base within expert
  const int nb = blockIdx.x * BN;   // f column base

  __shared__ __bf16 lx [BM * BK];   // X tile, [m][k] row-major    (16 KB)
  __shared__ __bf16 lwg[BK * BN];   // Wg tile, [n][k] row-major    (4 KB)
  __shared__ __bf16 lwu[BK * BN];   // Wu tile, [n][k] row-major    (4 KB)

  const float* xe = x   + (size_t)e * kTPE * kH;
  const float* wg = wgu + (size_t)e * 2 * kF * kH;   // gate rows [0,F)
  const float* wu = wg  + (size_t)kF * kH;           // up rows   [F,2F)

  const int tid  = threadIdx.x;
  const int wave = tid >> 5;
  const int lane = tid & 31;
  const int wm   = (wave >> 1) * WM;   // wave row offset in block tile
  const int wn   = (wave & 1)  * 32;   // wave col offset in block tile

  v8f accg[4][2] = {};
  v8f accu[4][2] = {};

  const int sk4  = (tid & 7) << 2;   // 4 consecutive k per thread (16B chunk)
  const int srow = tid >> 3;         // X staging: 32 rows per pass
  const int n2   = (tid >> 3) << 1;  // W staging: pair of n rows, one pass

  for (int kk = 0; kk < kH; kk += BK) {
    // ---- stage X tile: 16B global loads, packed f32 -> bf16 ----
    #pragma unroll
    for (int m = srow; m < BM; m += 32) {
      const float* gp = &xe[(size_t)(mb + m) * kH + kk + sk4];
      v4f v = *(const v4f*)gp;
      __builtin_prefetch(gp + BK, 0, 1);              // speculative, branch-free
      v4bf b = { (__bf16)v.x, (__bf16)v.y, (__bf16)v.z, (__bf16)v.w };
      *(v4bf*)&lx[m * BK + sk4] = b;
    }
    // ---- stage Wg/Wu as [n][k]: 4 consecutive-k bf16 per n row ----
    {
      const float* g0 = &wg[(size_t)(nb + n2) * kH + kk + sk4];
      const float* g1 = g0 + kH;
      const float* u0 = &wu[(size_t)(nb + n2) * kH + kk + sk4];
      const float* u1 = u0 + kH;
      v4f a0 = *(const v4f*)g0, a1 = *(const v4f*)g1;
      v4f b0 = *(const v4f*)u0, b1 = *(const v4f*)u1;
      __builtin_prefetch(g0 + BK, 0, 1);
      __builtin_prefetch(g1 + BK, 0, 1);
      __builtin_prefetch(u0 + BK, 0, 1);
      __builtin_prefetch(u1 + BK, 0, 1);
      *(v4bf*)&lwg[n2 * BK + sk4]       = (v4bf){ (__bf16)a0.x, (__bf16)a0.y, (__bf16)a0.z, (__bf16)a0.w };
      *(v4bf*)&lwg[(n2 + 1) * BK + sk4] = (v4bf){ (__bf16)a1.x, (__bf16)a1.y, (__bf16)a1.z, (__bf16)a1.w };
      *(v4bf*)&lwu[n2 * BK + sk4]       = (v4bf){ (__bf16)b0.x, (__bf16)b0.y, (__bf16)b0.z, (__bf16)b0.w };
      *(v4bf*)&lwu[(n2 + 1) * BK + sk4] = (v4bf){ (__bf16)b1.x, (__bf16)b1.y, (__bf16)b1.z, (__bf16)b1.w };
    }
    __syncthreads();

    // B fragments (32x16): N = lane&15, K in same order as A fragment.
    v16bf bg0 = load_a_frag(lwg, wn + (lane & 15), lane);
    v16bf bg1 = load_a_frag(lwg, wn + 16 + (lane & 15), lane);
    v16bf bu0 = load_a_frag(lwu, wn + (lane & 15), lane);
    v16bf bu1 = load_a_frag(lwu, wn + 16 + (lane & 15), lane);

    #pragma unroll
    for (int mi = 0; mi < 4; ++mi) {
      v16bf a = load_a_frag(lx, wm + mi * 16 + (lane & 15), lane);
      accg[mi][0] = __builtin_amdgcn_wmma_f32_16x16x32_bf16(false, a, false, bg0, (short)0, accg[mi][0], false, false);
      accg[mi][1] = __builtin_amdgcn_wmma_f32_16x16x32_bf16(false, a, false, bg1, (short)0, accg[mi][1], false, false);
      accu[mi][0] = __builtin_amdgcn_wmma_f32_16x16x32_bf16(false, a, false, bu0, (short)0, accu[mi][0], false, false);
      accu[mi][1] = __builtin_amdgcn_wmma_f32_16x16x32_bf16(false, a, false, bu1, (short)0, accu[mi][1], false, false);
    }
    __syncthreads();
  }

  // ---- epilogue: h = silu(g) * u, bf16 store ----
  // C/D layout: VGPR r -> M=r (lanes<16) or M=r+8 (lanes>=16); N = lane&15.
  __hip_bfloat16* hout = hidden + (size_t)e * kTPE * kF;
  const int ncol = nb + wn + (lane & 15);
  const int rofs = (lane >> 4) << 3;
  #pragma unroll
  for (int mi = 0; mi < 4; ++mi)
    #pragma unroll
    for (int ni = 0; ni < 2; ++ni) {
      const int nc = ncol + ni * 16;
      #pragma unroll
      for (int r = 0; r < 8; ++r) {
        const float hv = silu_fast(accg[mi][ni][r]) * accu[mi][ni][r];
        const int row = mb + wm + mi * 16 + rofs + r;
        hout[(size_t)row * kF + nc] = __float2bfloat16(hv);
      }
    }
}

// ---------------------------------------------------------------------------
// Kernel 2: out[e,t,h] = hidden @ Wd^T, f32 output.
// A tile staged with ASYNCcnt-tracked global_load_async_to_lds_b128.
// ---------------------------------------------------------------------------
__global__ __launch_bounds__(256) void gmlp_down(
    const __hip_bfloat16* __restrict__ hidden,  // [E, TPE, F] bf16
    const float* __restrict__ wdn,              // [E, H, F]
    float* __restrict__ out)                    // [E*TPE, H]
{
  const int e  = blockIdx.z;
  const int mb = blockIdx.y * BM;
  const int nb = blockIdx.x * BN;   // h column base

  __shared__ __bf16 la[BM * BK];    // hidden tile [m][k]           (16 KB)
  __shared__ __bf16 lw[BK * BN];    // Wd tile     [n][k] row-major  (4 KB)

  const __bf16* he = (const __bf16*)(hidden + (size_t)e * kTPE * kF);
  const float*  w  = wdn + (size_t)e * kH * kF;

  const int tid  = threadIdx.x;
  const int wave = tid >> 5;
  const int lane = tid & 31;
  const int wm   = (wave >> 1) * WM;
  const int wn   = (wave & 1)  * 32;

  v8f acc[4][2] = {};

  const int sk8  = (tid & 3) << 3;   // A staging: 8 bf16 (16B) per thread
  const int arow = tid >> 2;         // 64 rows per pass
  const int sk4  = (tid & 7) << 2;   // W staging: 4 f32 (16B) per thread
  const int n2   = (tid >> 3) << 1;  // W staging: pair of n rows, one pass

  for (int kk = 0; kk < kF; kk += BK) {
    // ---- A tile: async DMA global -> LDS (bf16 bytes, no conversion) ----
    #pragma unroll
    for (int m = arow; m < BM; m += 64) {
      const __bf16* gp = &he[(size_t)(mb + m) * kF + kk + sk8];
      unsigned lds_off = (unsigned)(size_t)&la[m * BK + sk8];
      asm volatile("global_load_async_to_lds_b128 %0, %1, off"
                   :: "v"(lds_off), "v"(gp) : "memory");
    }
    // ---- W tile as [n][k]: 4 consecutive-k bf16 per n row ----
    {
      const float* g0 = &w[(size_t)(nb + n2) * kF + kk + sk4];
      const float* g1 = g0 + kF;
      v4f a0 = *(const v4f*)g0, a1 = *(const v4f*)g1;
      __builtin_prefetch(g0 + BK, 0, 1);
      __builtin_prefetch(g1 + BK, 0, 1);
      *(v4bf*)&lw[n2 * BK + sk4]       = (v4bf){ (__bf16)a0.x, (__bf16)a0.y, (__bf16)a0.z, (__bf16)a0.w };
      *(v4bf*)&lw[(n2 + 1) * BK + sk4] = (v4bf){ (__bf16)a1.x, (__bf16)a1.y, (__bf16)a1.z, (__bf16)a1.w };
    }
    asm volatile("s_wait_asynccnt 0" ::: "memory");
    __syncthreads();

    v16bf b0 = load_a_frag(lw, wn + (lane & 15), lane);
    v16bf b1 = load_a_frag(lw, wn + 16 + (lane & 15), lane);

    #pragma unroll
    for (int mi = 0; mi < 4; ++mi) {
      v16bf a = load_a_frag(la, wm + mi * 16 + (lane & 15), lane);
      acc[mi][0] = __builtin_amdgcn_wmma_f32_16x16x32_bf16(false, a, false, b0, (short)0, acc[mi][0], false, false);
      acc[mi][1] = __builtin_amdgcn_wmma_f32_16x16x32_bf16(false, a, false, b1, (short)0, acc[mi][1], false, false);
    }
    __syncthreads();
  }

  float* oute = out + (size_t)e * kTPE * kH;
  const int ncol = nb + wn + (lane & 15);
  const int rofs = (lane >> 4) << 3;
  #pragma unroll
  for (int mi = 0; mi < 4; ++mi)
    #pragma unroll
    for (int ni = 0; ni < 2; ++ni) {
      const int nc = ncol + ni * 16;
      #pragma unroll
      for (int r = 0; r < 8; ++r) {
        const int row = mb + wm + mi * 16 + rofs + r;
        oute[(size_t)row * kH + nc] = acc[mi][ni][r];
      }
    }
}

extern "C" void kernel_launch(void* const* d_in, const int* in_sizes, int n_in,
                              void* d_out, int out_size, void* d_ws, size_t ws_size,
                              hipStream_t stream) {
  (void)in_sizes; (void)n_in; (void)out_size; (void)ws_size;
  const float* x   = (const float*)d_in[0];   // [T, H]
  const float* wgu = (const float*)d_in[1];   // [E, 2F, H]
  const float* wdn = (const float*)d_in[2];   // [E, H, F]
  // d_in[3] = tokens_per_expert (int64), constant T/E -> uniform partition.

  __hip_bfloat16* hidden = (__hip_bfloat16*)d_ws;  // E*TPE*F bf16 = 128 MiB
  float* out = (float*)d_out;

  dim3 blk(256);
  dim3 g1(kF / BN, kTPE / BM, kE);   // 64 x 8 x 8
  dim3 g2(kH / BN, kTPE / BM, kE);   // 32 x 8 x 8
  gmlp_gateup_silu<<<g1, blk, 0, stream>>>(x, wgu, hidden);
  gmlp_down<<<g2, blk, 0, stream>>>(hidden, wdn, out);
}
